// Attention_10307921511133
// MI455X (gfx1250) — compile-verified, matrix operand order checked by reading, not run on hardware
//
#include <hip/hip_runtime.h>

// ---------------------------------------------------------------------------
// Attention (B=8, Lq=Lk=2048, D=1024) for gfx1250 (MI455X), bf16 WMMA path.
//   1) cvt_bf16: fp32 -> bf16 (query, key, Wq_w, Wk_w)
//   2) vT:       value fp32 [B][Lk][D] -> bf16 [B][D][Lk] (transposed for B-frags)
//   3) proj:     q = X @ W^T + b (128x128 tiles, TDM-staged LDS, v_wmma bf16)
//   4) attn:     flash attention, 16 q-rows per WG, 4 waves x 256 d-cols,
//                online softmax, S/P shared via LDS, K chunk = 128 keys.
// ---------------------------------------------------------------------------

typedef unsigned short u16;
typedef unsigned int   u32;
typedef __attribute__((ext_vector_type(16))) __bf16 v16bf;
typedef __attribute__((ext_vector_type(8)))  float  v8f;
typedef __attribute__((ext_vector_type(4)))  u32    v4u;
typedef __attribute__((ext_vector_type(4)))  int    v4i;
typedef __attribute__((ext_vector_type(8)))  int    v8i;
typedef __attribute__((ext_vector_type(4)))  float  v4f;

#define D_DIM 1024
#define B_SZ  8
#define LQ_N  2048
#define LK_N  2048

// Toolchain probe: the gfx1250 TDM header only ships with the clang-23 /
// therock-10.0 stack, whose tensor_load_to_lds builtin takes 6 args.
#if __has_include(<hip/amd_detail/amd_gfx1250_TDM.h>)
#define TDM_SIX_ARG 1
#else
#define TDM_SIX_ARG 0
#endif

__device__ __forceinline__ u16 f2bf(float f) {
  u32 u = __float_as_uint(f);
  u32 r = (u + 0x7FFFu + ((u >> 16) & 1u)) >> 16;   // round-to-nearest-even
  return (u16)r;
}

union FragU { v16bf v; v4u q[2]; };

// Load a 16-element bf16 fragment slice: elements e<8 -> k = khalf*8+e,
// e>=8 -> k = 16 + khalf*8 + (e-8). `base` points at k=0 of the 32-wide K run.
__device__ __forceinline__ v16bf load_frag(const u16* base, int kh) {
  FragU f;
  f.q[0] = *reinterpret_cast<const v4u*>(base + kh * 8);
  f.q[1] = *reinterpret_cast<const v4u*>(base + 16 + kh * 8);
  return f.v;
}

__device__ __forceinline__ v8f wmma_bf16(v16bf a, v16bf b, v8f c) {
  return __builtin_amdgcn_wmma_f32_16x16x32_bf16(
      /*neg_a=*/false, a, /*neg_b=*/false, b,
      /*c_mod=*/(short)0, c, /*reuse_a=*/false, /*reuse_b=*/false);
}

// 2D TDM load: tile_d0 x tile_d1 elements (2 bytes each), global row stride
// `stride0` elements, into contiguous LDS at byte offset lds_off.
// D# bitfields per CDNA5 ISA ch.8 (group0: count/lds/global/type,
// group1: data_size, dims, tile dims, strides). Groups 2/3 zero (2D tile).
__device__ __forceinline__ void tdm_load_2d(u32 lds_off, const void* gptr,
                                            u32 tile_d0, u32 tile_d1,
                                            u32 tensor_d0, u32 tensor_d1,
                                            u32 stride0) {
  unsigned long long ga = (unsigned long long)(size_t)gptr;
  v4u g0;
  g0.x = 1u;                                         // count=1 (valid user D#)
  g0.y = lds_off;                                    // lds_addr[31:0]
  g0.z = (u32)(ga & 0xFFFFFFFFu);                    // global_addr[31:0]
  g0.w = (u32)((ga >> 32) & 0x1FFFFFFu) | (2u << 30); // global_addr[56:32] | type=2
  v8i g1;
  g1[0] = (int)(1u << 16);                           // wg_mask=0, data_size=1 (2B)
  g1[1] = (int)((tensor_d0 & 0xFFFFu) << 16);        // tensor_dim0[15:0] @63:48
  g1[2] = (int)(((tensor_d0 >> 16) & 0xFFFFu) |      // tensor_dim0[31:16]
                ((tensor_d1 & 0xFFFFu) << 16));      // tensor_dim1[15:0]
  g1[3] = (int)(((tensor_d1 >> 16) & 0xFFFFu) |      // tensor_dim1[31:16]
                ((tile_d0 & 0xFFFFu) << 16));        // tile_dim0 @127:112
  g1[4] = (int)(tile_d1 & 0xFFFFu);                  // tile_dim1; tile_dim2=0
  g1[5] = (int)stride0;                              // tensor_dim0_stride[31:0]
  g1[6] = 0;                                         // stride0[47:32] | stride1[15:0]
  g1[7] = 0;                                         // stride1[47:16]
  v4i z4 = {0, 0, 0, 0};
#if TDM_SIX_ARG
  v8i z8 = {0, 0, 0, 0, 0, 0, 0, 0};
  __builtin_amdgcn_tensor_load_to_lds(g0, g1, z4, z4, z8, 0);
#else
  __builtin_amdgcn_tensor_load_to_lds(g0, g1, z4, z4, 0);
#endif
}

// ---------------------------------------------------------------- cvt fp32->bf16
__global__ __launch_bounds__(256) void cvt_bf16_kernel(
    const float* __restrict__ in, u16* __restrict__ out, int n) {
  int i = (blockIdx.x * 256 + threadIdx.x) * 8;
  if (i + 8 > n) return;
  v4f a = *reinterpret_cast<const v4f*>(in + i);
  v4f b = *reinterpret_cast<const v4f*>(in + i + 4);
  union { u16 s[8]; v4u q; } o;
  o.s[0] = f2bf(a.x); o.s[1] = f2bf(a.y); o.s[2] = f2bf(a.z); o.s[3] = f2bf(a.w);
  o.s[4] = f2bf(b.x); o.s[5] = f2bf(b.y); o.s[6] = f2bf(b.z); o.s[7] = f2bf(b.w);
  *reinterpret_cast<v4u*>(out + i) = o.q;
}

// ------------------------------------------------- value -> bf16 [B][D][Lk]
__global__ __launch_bounds__(256) void vT_kernel(
    const float* __restrict__ v, u16* __restrict__ vt) {
  int i = blockIdx.x * 256 + threadIdx.x;            // < B*Lk*D
  int d = i & (D_DIM - 1);
  int rest = i >> 10;                                 // b*Lk + k
  int k = rest & (LK_N - 1);
  int b = rest >> 11;
  vt[((size_t)(b * D_DIM + d) << 11) + k] = f2bf(v[i]);
}

// ------------------------------------------------------- projection GEMM
// O[M,1024] = X[M,1024] @ W^T + bias, M = B*Lq = 16384, bf16 in/out, f32 acc.
// WG = 256 thr (8 waves), tile 128x128, wave = 64x32 (4x2 wmma tiles).
// A/B panels staged to LDS by the Tensor Data Mover (wave 0 issues, waits on
// TENSORcnt, then the workgroup barrier publishes the LDS tile).
__global__ __launch_bounds__(256) void proj_kernel(
    const u16* __restrict__ Xq, const u16* __restrict__ Xk,
    const u16* __restrict__ Wq, const u16* __restrict__ Wk,
    const float* __restrict__ bq, const float* __restrict__ bk,
    u16* __restrict__ Oq, u16* __restrict__ Ok) {
  __shared__ __align__(16) u16 Als[128 * 32];
  __shared__ __align__(16) u16 Bls[128 * 32];

  const bool isK = (blockIdx.z != 0);
  const u16*  X    = isK ? Xk : Xq;
  const u16*  W    = isK ? Wk : Wq;
  const float* bias = isK ? bk : bq;
  u16*        O    = isK ? Ok : Oq;

  const int m0 = blockIdx.x * 128;
  const int n0 = blockIdx.y * 128;
  const int tid  = threadIdx.x;
  const int lane = tid & 31;
  const int wid  = tid >> 5;
  const int lr   = lane & 15;
  const int kh   = lane >> 4;
  const int wm   = (wid >> 2) * 64;   // wave m offset (0 or 64)
  const int wn   = (wid & 3) * 32;    // wave n offset (0..96)

  const u32 ldsA = (u32)(size_t)&Als[0];
  const u32 ldsB = (u32)(size_t)&Bls[0];

  v8f acc[4][2];
  v8f zero = {};
#pragma unroll
  for (int i = 0; i < 4; ++i)
#pragma unroll
    for (int j = 0; j < 2; ++j) acc[i][j] = zero;

  for (int k0 = 0; k0 < D_DIM; k0 += 32) {
    if (wid == 0) {   // one wave drives the TDM for both panels
      tdm_load_2d(ldsA, X + (size_t)m0 * D_DIM + k0, 32, 128, 32, 128, D_DIM);
      tdm_load_2d(ldsB, W + (size_t)n0 * D_DIM + k0, 32, 128, 32, 128, D_DIM);
      __builtin_amdgcn_s_wait_tensorcnt(0);
    }
    __syncthreads();

    v16bf af[4], bf[2];
#pragma unroll
    for (int i = 0; i < 4; ++i)
      af[i] = load_frag(Als + (wm + i * 16 + lr) * 32, kh);
#pragma unroll
    for (int j = 0; j < 2; ++j)
      bf[j] = load_frag(Bls + (wn + j * 16 + lr) * 32, kh);
#pragma unroll
    for (int i = 0; i < 4; ++i)
#pragma unroll
      for (int j = 0; j < 2; ++j)
        acc[i][j] = wmma_bf16(af[i], bf[j], acc[i][j]);
    __syncthreads();
  }

#pragma unroll
  for (int i = 0; i < 4; ++i)
#pragma unroll
    for (int j = 0; j < 2; ++j) {
      int col = n0 + wn + j * 16 + lr;
      float bv = bias[col];
#pragma unroll
      for (int r = 0; r < 8; ++r) {
        int row = m0 + wm + i * 16 + r + 8 * kh;   // C layout: VGPR r -> M=r (+8 hi half)
        O[(size_t)row * D_DIM + col] = f2bf(acc[i][j][r] + bv);
      }
    }
}

// -------------------------------------------------------------- attention
// WG = 128 thr (4 waves). 16 q-rows per WG. Wave w owns d-cols [256w, 256w+256).
// Loop over Lk in chunks of 128 keys; wave w computes S for its 32 keys.
__global__ __launch_bounds__(128) void attn_kernel(
    const u16* __restrict__ qb, const u16* __restrict__ kb,
    const u16* __restrict__ vt, const int* __restrict__ mask,
    float* __restrict__ out) {
  __shared__ __align__(16) u16   Qs[16 * D_DIM];   // 32 KB
  __shared__ __align__(16) float Sls[16 * 128];    //  8 KB
  __shared__ __align__(16) u16   Pls[16 * 128];    //  4 KB

  const int b   = blockIdx.y;
  const int q0  = blockIdx.x * 16;
  const int tid = threadIdx.x;
  const int lane = tid & 31;
  const int wid  = tid >> 5;    // 0..3
  const int lr   = lane & 15;
  const int kh   = lane >> 4;
  const float scale = 0.03125f; // 1/sqrt(1024)

  { // stage Q tile (16 x 1024 bf16) into LDS
    int row = tid >> 3;
    int dbase = (tid & 7) * 128;
    const v4u* src = reinterpret_cast<const v4u*>(
        qb + (size_t)(b * LQ_N + q0 + row) * D_DIM + dbase);
    v4u* dst = reinterpret_cast<v4u*>(Qs + row * D_DIM + dbase);
#pragma unroll
    for (int j = 0; j < 16; ++j) dst[j] = src[j];
  }
  __syncthreads();

  v8f accO[16];
  v8f zero = {};
#pragma unroll
  for (int j = 0; j < 16; ++j) accO[j] = zero;

  float m_i = -1e30f, l_i = 0.0f;   // online-softmax stats for row lr

  for (int k0 = 0; k0 < LK_N; k0 += 128) {
    if (k0 + 128 < LK_N)
      __builtin_prefetch(kb + (size_t)(b * LK_N + k0 + 128 + lane) * D_DIM, 0, 1);

    // ---- Phase A: S tile for this wave's 32 keys, contracted over D=1024
    v8f c0 = zero, c1 = zero;
    const u16* qrow  = Qs + lr * D_DIM;
    const u16* krow0 = kb + (size_t)(b * LK_N + k0 + wid * 32 + lr) * D_DIM;
    const u16* krow1 = krow0 + (size_t)16 * D_DIM;
    for (int d0 = 0; d0 < D_DIM; d0 += 32) {
      v16bf aq = load_frag(qrow + d0, kh);
      v16bf b0 = load_frag(krow0 + d0, kh);
      v16bf b1 = load_frag(krow1 + d0, kh);
      c0 = wmma_bf16(aq, b0, c0);
      c1 = wmma_bf16(aq, b1, c1);
    }
    float sreg[16];
#pragma unroll
    for (int t = 0; t < 2; ++t)
#pragma unroll
      for (int r = 0; r < 8; ++r) {
        int row  = r + 8 * kh;
        int keyl = wid * 32 + t * 16 + lr;
        int mv   = mask[b * LK_N + k0 + keyl];
        float s  = (t == 0) ? c0[r] : c1[r];
        s = mv ? s * scale : -1e9f;     // aw*m + where(m==0, NEG)
        sreg[t * 8 + r] = s;
        Sls[row * 128 + keyl] = s;
      }
    __syncthreads();

    // ---- Phase B: row stats (lane handles row lr; both halves redundant)
    float mx = -1e30f;
    for (int k = 0; k < 128; ++k) mx = fmaxf(mx, Sls[lr * 128 + k]);
    float m_new = fmaxf(m_i, mx);
    float corr  = __expf(m_i - m_new);
    float ssum  = 0.0f;
    for (int k = 0; k < 128; ++k) ssum += __expf(Sls[lr * 128 + k] - m_new);
    l_i = l_i * corr + ssum;
    m_i = m_new;

#pragma unroll
    for (int t = 0; t < 2; ++t)
#pragma unroll
      for (int r = 0; r < 8; ++r) {
        int row = r + 8 * kh;
        float mr = __shfl(m_i, row, 32);
        float p  = __expf(sreg[t * 8 + r] - mr);
        Pls[row * 128 + wid * 32 + t * 16 + lr] = f2bf(p);
      }
    __syncthreads();

    // ---- Phase C: rescale O, accumulate P @ V for this wave's 256 d-cols
    float crr[8];
#pragma unroll
    for (int r = 0; r < 8; ++r) crr[r] = __shfl(corr, r + 8 * kh, 32);
#pragma unroll
    for (int j = 0; j < 16; ++j)
#pragma unroll
      for (int r = 0; r < 8; ++r) accO[j][r] *= crr[r];

    for (int kk = 0; kk < 4; ++kk) {
      v16bf pf = load_frag(Pls + lr * 128 + kk * 32, kh);
#pragma unroll
      for (int j = 0; j < 16; ++j) {
        int dcol = wid * 256 + j * 16 + lr;
        const u16* vrow = vt + (size_t)(b * D_DIM + dcol) * LK_N + k0 + kk * 32;
        v16bf vf = load_frag(vrow, kh);
        accO[j] = wmma_bf16(pf, vf, accO[j]);
      }
    }
  }

  // ---- epilogue: O / l, f32 store
  float lrow[8];
#pragma unroll
  for (int r = 0; r < 8; ++r) lrow[r] = __shfl(l_i, r + 8 * kh, 32);
#pragma unroll
  for (int r = 0; r < 8; ++r) {
    int row = q0 + r + 8 * kh;
    float inv = 1.0f / lrow[r];
#pragma unroll
    for (int j = 0; j < 16; ++j) {
      int dcol = wid * 256 + j * 16 + lr;
      out[((size_t)b * LQ_N + row) * D_DIM + dcol] = accO[j][r] * inv;
    }
  }
}

// ---------------------------------------------------------------------------
extern "C" void kernel_launch(void* const* d_in, const int* in_sizes, int n_in,
                              void* d_out, int out_size, void* d_ws, size_t ws_size,
                              hipStream_t stream) {
  const float* query = (const float*)d_in[0];
  const float* key   = (const float*)d_in[1];
  const float* value = (const float*)d_in[2];
  const int*   mask  = (const int*)d_in[3];
  const float* Wq_w  = (const float*)d_in[4];
  const float* Wq_b  = (const float*)d_in[5];
  const float* Wk_w  = (const float*)d_in[6];
  const float* Wk_b  = (const float*)d_in[7];
  float* out = (float*)d_out;

  char* ws = (char*)d_ws;
  const size_t SZ_X = (size_t)B_SZ * LQ_N * D_DIM * sizeof(u16);  // 33.5 MB
  const size_t SZ_W = (size_t)D_DIM * D_DIM * sizeof(u16);        //  2  MB
  u16* queryb = (u16*)ws;            ws += SZ_X;
  u16* keyb   = (u16*)ws;            ws += SZ_X;
  u16* Wqb    = (u16*)ws;            ws += SZ_W;
  u16* Wkb    = (u16*)ws;            ws += SZ_W;
  u16* qbuf   = (u16*)ws;            ws += SZ_X;
  u16* kbuf   = (u16*)ws;            ws += SZ_X;
  u16* vtbuf  = (u16*)ws;            ws += SZ_X;

  const int nX = B_SZ * LQ_N * D_DIM;   // 16,777,216
  const int nW = D_DIM * D_DIM;         //  1,048,576

  cvt_bf16_kernel<<<nX / 2048, 256, 0, stream>>>(query, queryb, nX);
  cvt_bf16_kernel<<<nX / 2048, 256, 0, stream>>>(key,   keyb,   nX);
  cvt_bf16_kernel<<<nW / 2048, 256, 0, stream>>>(Wq_w,  Wqb,    nW);
  cvt_bf16_kernel<<<nW / 2048, 256, 0, stream>>>(Wk_w,  Wkb,    nW);
  vT_kernel<<<nX / 256, 256, 0, stream>>>(value, vtbuf);

  proj_kernel<<<dim3((B_SZ * LQ_N) / 128, D_DIM / 128, 2), 256, 0, stream>>>(
      queryb, keyb, Wqb, Wkb, Wq_b, Wk_b, qbuf, kbuf);

  attn_kernel<<<dim3(LQ_N / 16, B_SZ), 128, 0, stream>>>(
      qbuf, kbuf, vtbuf, mask, out);
}
